// GGAN_82995948028454
// MI455X (gfx1250) — compile-verified, matrix operand order checked by reading, not run on hardware
//
#include <hip/hip_runtime.h>
#include <hip/hip_bf16.h>

// Problem constants (match reference)
#define B_   4
#define N_   2048
#define FIN_ 256
#define D_   128
#define H_   4

typedef __attribute__((ext_vector_type(16))) __bf16 v16bf;
typedef __attribute__((ext_vector_type(8)))  __bf16 v8bf;
typedef __attribute__((ext_vector_type(4)))  __bf16 v4bf;
typedef __attribute__((ext_vector_type(8)))  float  v8f;
typedef __attribute__((ext_vector_type(4)))  float  v4f;

__device__ __forceinline__ v8f wmma_bf16(v16bf a, v16bf b, v8f c) {
    // (neg_a, A, neg_b, B, c_mod, C, reuse_a, reuse_b)
    return __builtin_amdgcn_wmma_f32_16x16x32_bf16(false, a, false, b, (short)0, c, false, false);
}

// Fragment load for one 32-wide K slab, 16-bit data (per ISA 7.12.2 layout):
// lane-half 0 holds K 0..7 then 16..23, lane-half 1 holds K 8..15 then 24..31.
// Both 8-element groups are contiguous bf16 runs -> two ds_load_b128.
__device__ __forceinline__ v16bf ldfrag(const __bf16* p, int half) {
    const v8bf lo = *reinterpret_cast<const v8bf*>(p + half * 8);
    const v8bf hi = *reinterpret_cast<const v8bf*>(p + 16 + half * 8);
    return __builtin_shufflevector(lo, hi, 0, 1, 2, 3, 4, 5, 6, 7,
                                           8, 9, 10, 11, 12, 13, 14, 15);
}

__device__ __forceinline__ v4bf cvt4(v4f x) {
    v4bf r;
    r[0] = (__bf16)x[0]; r[1] = (__bf16)x[1];
    r[2] = (__bf16)x[2]; r[3] = (__bf16)x[3];
    return r;
}

// --- CDNA5 async copy path (ISA 15.18.3: GLOBAL_LOAD_ASYNC_TO_LDS_B128) ----
// VDST = LDS byte address VGPR, VADDR = 64-bit global address, saddr = off.
// Generic LDS pointers hold the DS offset in the low 32 bits (aperture rule),
// so truncation yields the LDS address. Tracked by ASYNCcnt.
__device__ __forceinline__ unsigned ldsaddr(const void* p) {
    return (unsigned)(unsigned long)p;
}
__device__ __forceinline__ void async_b128(const void* g, unsigned lds) {
    asm volatile("global_load_async_to_lds_b128 %0, %1, off"
                 :: "v"(lds), "v"(g) : "memory");
}
__device__ __forceinline__ void wait_async0() {
    asm volatile("s_wait_asynccnt 0x0" ::: "memory");
}

// ---------------------------------------------------------------------------
// Generic tiled GEMM: C[M x 128] = A[M x K] * B[K x 128] (+bias), bf16 WMMA.
// Block = 256 threads (8 waves). Block tile = 64 rows x 128 cols.
// B tile stored TRANSPOSED in LDS (sBt[col][k]) -> contiguous b128 fragments.
// Optional outputs: f32 (Cp) and/or bf16 (Cbf).
// Batch: z = blockIdx.z; outer = z / zdiv; inner = z % zdiv.
// BT=true reads global B as B[n*ldb + k] (i.e. B stored [n, k] row-major).
// ---------------------------------------------------------------------------
template <bool BT>
__global__ __launch_bounds__(256)
void gemm_wmma(const float* __restrict__ Ap, const float* __restrict__ Bp,
               const float* __restrict__ bias,
               float* __restrict__ Cp, __bf16* __restrict__ Cbf,
               int K, int lda, int ldb,
               int zdiv, long aOut, long aIn, long bOut, long bIn,
               long cB, long cbB)
{
    const int z = blockIdx.z;
    const int outer = z / zdiv, inner = z % zdiv;
    Ap += outer * aOut + inner * aIn;
    Bp += outer * bOut + inner * bIn;
    if (Cp)  Cp  += (long)z * cB;
    if (Cbf) Cbf += (long)z * cbB;
    const int m0 = blockIdx.x * 64;

    __shared__ __align__(16) __bf16 sA[64][40];    // [row][k] (+pad, 80B rows)
    __shared__ __align__(16) __bf16 sBt[128][40];  // [col][k] transposed

    const int tid  = threadIdx.x;
    const int lane = tid & 31;
    const int w    = tid >> 5;
    const int half = lane >> 4;
    const int l16  = lane & 15;
    const int wr   = (w >> 1) * 16;  // wave row offset inside block tile
    const int wc   = (w & 1) * 64;   // wave col offset inside block tile

    // Per-thread staging bases (strength-reduced addressing)
    const float* gA0 = Ap + (long)(m0 + (tid >> 3)) * lda + (tid & 7) * 4;   // A quads
    const long  aStep = 32l * lda;
    __bf16* lA0 = &sA[tid >> 3][(tid & 7) * 4];

    const float* gBt0 = BT ? (Bp + (long)(tid >> 3) * ldb + (tid & 7) * 4)   // [col][k]
                           : (Bp + (long)(tid >> 5) * ldb + (tid & 31) * 4); // [k][col]
    const long  bStep = BT ? 32l * ldb : 8l * ldb;

    v8f acc[4] = {};

    for (int k0 = 0; k0 < K; k0 += 32) {
        // Stage A tile: 64x32 = 512 float4, 2 per thread
        #pragma unroll
        for (int j = 0; j < 2; j++) {
            v4f v = *reinterpret_cast<const v4f*>(gA0 + k0 + j * aStep);
            *reinterpret_cast<v4bf*>(lA0 + j * 32 * 40) = cvt4(v);
        }
        // Stage B tile transposed into sBt[col][k]
        if (BT) {
            // global [col][k]: contiguous in k -> b128 load + b64 LDS store
            __bf16* lB0 = &sBt[tid >> 3][(tid & 7) * 4];
            #pragma unroll
            for (int j = 0; j < 4; j++) {
                v4f v = *reinterpret_cast<const v4f*>(gBt0 + k0 + j * bStep);
                *reinterpret_cast<v4bf*>(lB0 + j * 32 * 40) = cvt4(v);
            }
        } else {
            // global [k][col]: coalesced b128 along col, scatter 4 b16 stores
            const int q = (tid & 31) * 4, r0 = tid >> 5;
            #pragma unroll
            for (int j = 0; j < 4; j++) {
                v4f v = *reinterpret_cast<const v4f*>(gBt0 + (long)k0 * ldb + j * bStep);
                v4bf c = cvt4(v);
                int r = r0 + 8 * j;
                sBt[q + 0][r] = c[0];
                sBt[q + 1][r] = c[1];
                sBt[q + 2][r] = c[2];
                sBt[q + 3][r] = c[3];
            }
        }
        __syncthreads();

        v16bf af = ldfrag(&sA[wr + l16][0], half);
        #pragma unroll
        for (int ct = 0; ct < 4; ct++) {
            v16bf bfr = ldfrag(&sBt[wc + ct * 16 + l16][0], half);
            acc[ct] = wmma_bf16(af, bfr, acc[ct]);
        }
        __syncthreads();
    }

    // Epilogue: C layout = lane gives col (l16), VGPR v gives row v + 8*half
    if (Cp) {
        #pragma unroll
        for (int ct = 0; ct < 4; ct++) {
            int col = wc + ct * 16 + l16;
            float bv = bias ? bias[col] : 0.0f;
            #pragma unroll
            for (int v = 0; v < 8; v++)
                Cp[(long)(m0 + wr + v + 8 * half) * D_ + col] = acc[ct][v] + bv;
        }
    }
    if (Cbf) {
        #pragma unroll
        for (int ct = 0; ct < 4; ct++) {
            int col = wc + ct * 16 + l16;
            float bv = bias ? bias[col] : 0.0f;
            #pragma unroll
            for (int v = 0; v < 8; v++)
                Cbf[(long)(m0 + wr + v + 8 * half) * D_ + col] =
                    (__bf16)(acc[ct][v] + bv);
        }
    }
}

// ---------------------------------------------------------------------------
// Fused masked attention:
//   heads[h,b,n,d] = sum_m tanh(A[b,n,m] * (Hx[b,n,:] . Pt[b,h,m,:])) * Hx[b,m,d]
//   Hout = relu(mean_h heads)
// S/Am ([H,B,N,N] = 268MB) never materialized; A stays L2-resident (64MB<192MB).
// Hx/Pt consumed as bf16; pure-copy tiles use global_load_async_to_lds_b128
// with a double-buffered Pt tile pipelined across the head loop (ASYNCcnt).
// Block = 128 threads (4 waves); block tile = 64 n-rows; each wave owns 16.
// ---------------------------------------------------------------------------
__global__ __launch_bounds__(128)
void attn_fused(const float* __restrict__ hx,     // [B,N,D] f32 (transpose src)
                const __bf16* __restrict__ hxbf,  // [B,N,D] bf16
                const __bf16* __restrict__ ptbf,  // [B,H,N,D] bf16
                const float* __restrict__ Aadj,   // [B,N,N]
                float* __restrict__ hout)         // [B,N,D]
{
    const int b  = blockIdx.y;
    const int n0 = blockIdx.x * 64;

    __shared__ __align__(16) __bf16 sHn[64][136];     // [n][d] A-op for S
    __shared__ __align__(16) __bf16 sHmT[128][40];    // [d][m] B-op for heads
    __shared__ __align__(16) __bf16 sPt[2][32][136];  // [m][d] B-op for S (2-buf)
    __shared__ __align__(16) float  sAdj[64][36];     // A[n,m] tile
    __shared__ __align__(16) __bf16 sAm[4][16][40];   // per-wave Am restage

    const int tid  = threadIdx.x;
    const int lane = tid & 31;
    const int w    = tid >> 5;
    const int half = lane >> 4;
    const int l16  = lane & 15;

    // Per-thread staging bases
    const __bf16* gHn0 = hxbf + (long)b * N_ * D_ +
                         (long)(n0 + (tid >> 4)) * D_ + (tid & 15) * 8;
    const __bf16* gPt0 = ptbf + (long)b * H_ * N_ * D_ +
                         (long)(tid >> 4) * D_ + (tid & 15) * 8;
    const float*  gHm0 = hx + (long)b * N_ * D_ +
                         (long)(tid >> 5) * D_ + (tid & 31) * 4;
    const float*  gAd0 = Aadj + (long)b * N_ * N_ +
                         (long)(n0 + (tid >> 3)) * N_ + (tid & 7) * 4;

    const unsigned lHn  = ldsaddr(&sHn[tid >> 4][(tid & 15) * 8]);
    const unsigned lPt0 = ldsaddr(&sPt[0][tid >> 4][(tid & 15) * 8]);
    const unsigned lPt1 = ldsaddr(&sPt[1][tid >> 4][(tid & 15) * 8]);
    const unsigned lAd  = ldsaddr(&sAdj[tid >> 3][(tid & 7) * 4]);

    // Stage this block's 64 Hx rows once: async b128 copy, 8 per thread
    #pragma unroll
    for (int j = 0; j < 8; j++)
        async_b128(gHn0 + j * 8 * D_, lHn + j * 8 * 136 * 2);

    v8f acc[8] = {};   // 16 rows x 128 cols of head accumulation (sum over h)

    for (int m0 = 0; m0 < N_; m0 += 32) {
        // Async: adjacency tile (f32) + Pt tile for h=0 into buffer 0
        #pragma unroll
        for (int j = 0; j < 4; j++)
            async_b128(gAd0 + m0 + (long)j * 16 * N_, lAd + j * 16 * 36 * 4);
        {
            const __bf16* g = gPt0 + (long)m0 * D_;
            #pragma unroll
            for (int j = 0; j < 4; j++)
                async_b128(g + j * 8 * D_, lPt0 + j * 8 * 136 * 2);
        }
        if (m0 + 32 < N_)
            __builtin_prefetch(gHm0 + (long)(m0 + 32) * D_, 0, 0);

        // Hx m-tile transposed (from f32 copy): 8 quads per thread (VALU cvt)
        {
            const float* g = gHm0 + (long)m0 * D_;
            const int q = (tid & 31) * 4, r0 = tid >> 5;
            #pragma unroll
            for (int j = 0; j < 8; j++) {
                v4f v = *reinterpret_cast<const v4f*>(g + j * 4 * D_);
                v4bf c = cvt4(v);
                int r = r0 + 4 * j;
                sHmT[q + 0][r] = c[0];
                sHmT[q + 1][r] = c[1];
                sHmT[q + 2][r] = c[2];
                sHmT[q + 3][r] = c[3];
            }
        }
        wait_async0();
        __syncthreads();

        for (int h = 0; h < H_; h++) {
            const int p = h & 1;
            // Pipeline: issue next head's Pt tile into the other buffer
            if (h + 1 < H_) {
                const __bf16* g = gPt0 + (long)(h + 1) * N_ * D_ + (long)m0 * D_;
                const unsigned lnext = (p == 0) ? lPt1 : lPt0;
                #pragma unroll
                for (int j = 0; j < 4; j++)
                    async_b128(g + j * 8 * D_, lnext + j * 8 * 136 * 2);
            }

            // S tiles: [16 n-rows] x [32 m-cols], contracted over d (K=128)
            v8f s0 = {}, s1 = {};
            #pragma unroll
            for (int d0 = 0; d0 < D_; d0 += 32) {
                v16bf af = ldfrag(&sHn[w * 16 + l16][d0], half);
                v16bf b0 = ldfrag(&sPt[p][l16][d0], half);      // m cols 0..15
                v16bf b1 = ldfrag(&sPt[p][16 + l16][d0], half); // m cols 16..31
                s0 = wmma_bf16(af, b0, s0);
                s1 = wmma_bf16(af, b1, s1);
            }

            // Am = tanh(A * S); restage from C layout into A-fragment layout
            #pragma unroll
            for (int v = 0; v < 8; v++) {
                int rl = v + 8 * half;                      // row in wave tile
                float a0 = sAdj[w * 16 + rl][l16];
                float a1 = sAdj[w * 16 + rl][16 + l16];
                sAm[w][rl][l16]      = (__bf16)tanhf(a0 * s0[v]);
                sAm[w][rl][16 + l16] = (__bf16)tanhf(a1 * s1[v]);
            }
            // wave-private LDS roundtrip: only need DS counter drained
            asm volatile("s_wait_dscnt 0x0" ::: "memory");

            v16bf amf = ldfrag(&sAm[w][l16][0], half);

            // heads += Am(16x32) @ Hx_m(32x128)
            #pragma unroll
            for (int dt = 0; dt < 8; dt++) {
                v16bf bfr = ldfrag(&sHmT[dt * 16 + l16][0], half);
                acc[dt] = wmma_bf16(amf, bfr, acc[dt]);
            }

            if (h + 1 < H_) {
                wait_async0();      // next head's Pt tile landed
                __syncthreads();    // ...for every wave
            }
        }
        __syncthreads();            // retire sHmT/sAdj/sPt reads before restage
    }

    // Hout = relu(mean over heads)
    float* ho = hout + (long)b * N_ * D_;
    #pragma unroll
    for (int dt = 0; dt < 8; dt++) {
        #pragma unroll
        for (int v = 0; v < 8; v++) {
            int row = n0 + w * 16 + v + 8 * half;
            int col = dt * 16 + l16;
            float val = acc[dt][v] * (1.0f / H_);
            ho[(long)row * D_ + col] = fmaxf(val, 0.0f);
        }
    }
}

// ---------------------------------------------------------------------------
// Final gate: z = sigmoid(g1 + g2 + update_bias); out = Hout*z + Xs*(1-z)
// ---------------------------------------------------------------------------
__global__ __launch_bounds__(256)
void gate_out(const float* __restrict__ hout, const float* __restrict__ xs,
              const float* __restrict__ g1, const float* __restrict__ g2,
              const float* __restrict__ ub, float* __restrict__ out, int nquads)
{
    int i = blockIdx.x * blockDim.x + threadIdx.x;
    if (i < nquads) {
        long i4 = (long)i * 4;
        v4f a = *reinterpret_cast<const v4f*>(&g1[i4]);
        v4f bq = *reinterpret_cast<const v4f*>(&g2[i4]);
        v4f u = *reinterpret_cast<const v4f*>(&ub[i4 & (D_ - 1)]);
        v4f h = *reinterpret_cast<const v4f*>(&hout[i4]);
        v4f x = *reinterpret_cast<const v4f*>(&xs[i4]);
        v4f o;
        #pragma unroll
        for (int c = 0; c < 4; c++) {
            float zg = 1.0f / (1.0f + __expf(-(a[c] + bq[c] + u[c])));
            o[c] = h[c] * zg + x[c] * (1.0f - zg);
        }
        *reinterpret_cast<v4f*>(&out[i4]) = o;
    }
}

extern "C" void kernel_launch(void* const* d_in, const int* in_sizes, int n_in,
                              void* d_out, int out_size, void* d_ws, size_t ws_size,
                              hipStream_t stream)
{
    const float* X       = (const float*)d_in[0];   // [B,N,FIN]
    const float* Aadj    = (const float*)d_in[1];   // [B,N,N]
    const float* kernelW = (const float*)d_in[2];   // [FIN,D]
    const float* bias    = (const float*)d_in[3];   // [D]
    const float* ubias   = (const float*)d_in[4];   // [D]
    const float* attn    = (const float*)d_in[5];   // [H,D,D]
    const float* W_skip  = (const float*)d_in[6];   // [FIN,D]
    const float* b_skip  = (const float*)d_in[7];   // [D]
    const float* W_u     = (const float*)d_in[8];   // [D,D]
    const float* b_u     = (const float*)d_in[9];   // [D]
    const float* W_x     = (const float*)d_in[10];  // [D,D]
    const float* b_x     = (const float*)d_in[11];  // [D]

    const long BND = (long)B_ * N_ * D_;            // 1,048,576
    float* ws   = (float*)d_ws;
    float* hx0  = ws + 0 * BND;
    float* hx   = ws + 1 * BND;
    float* xs   = ws + 2 * BND;
    float* hout = ws + 3 * BND;
    float* g1   = ws + 4 * BND;
    float* g2   = ws + 5 * BND;
    __bf16* hxbf = (__bf16*)(ws + 6 * BND);         // [B,N,D] bf16
    __bf16* ptbf = hxbf + BND;                      // [B,H,N,D] bf16 (4*BND)

    const int  MTOT = B_ * N_;                      // 8192
    const long ND = (long)N_ * D_, NN = (long)N_ * N_, DD = (long)D_ * D_;

    // 1) Hx0 = X @ kernel + bias          (M=8192, K=256)
    gemm_wmma<false><<<dim3(MTOT / 64, 1, 1), 256, 0, stream>>>(
        X, kernelW, bias, hx0, nullptr, FIN_, FIN_, D_, 1, 0, 0, 0, 0, 0, 0);

    // 2) Xs = X @ W_skip + b_skip         (M=8192, K=256)
    gemm_wmma<false><<<dim3(MTOT / 64, 1, 1), 256, 0, stream>>>(
        X, W_skip, b_skip, xs, nullptr, FIN_, FIN_, D_, 1, 0, 0, 0, 0, 0, 0);

    // 3) Hx = A @ Hx0 per batch           (M=2048, K=2048, z=B); f32 + bf16 out
    gemm_wmma<false><<<dim3(N_ / 64, 1, B_), 256, 0, stream>>>(
        Aadj, hx0, nullptr, hx, hxbf, N_, N_, D_,
        1, NN, 0, ND, 0, ND, ND);

    // 4) Pt[b,h,m,d] = Hx[b,m,:] @ attn[h,:,:]^T  (M=2048, K=128, z=B*H); bf16
    gemm_wmma<true><<<dim3(N_ / 64, 1, B_ * H_), 256, 0, stream>>>(
        hx, attn, nullptr, nullptr, ptbf, D_, D_, D_,
        H_, ND, 0, 0, DD, 0, ND);

    // 5) Fused masked attention + head mean + relu -> Hout
    attn_fused<<<dim3(N_ / 64, B_, 1), 128, 0, stream>>>(
        hx, hxbf, ptbf, Aadj, hout);

    // 6) g1 = Hout @ W_u + b_u ; g2 = Xs @ W_x + b_x   (M=8192, K=128)
    gemm_wmma<false><<<dim3(MTOT / 64, 1, 1), 256, 0, stream>>>(
        hout, W_u, b_u, g1, nullptr, D_, D_, D_, 1, 0, 0, 0, 0, 0, 0);
    gemm_wmma<false><<<dim3(MTOT / 64, 1, 1), 256, 0, stream>>>(
        xs, W_x, b_x, g2, nullptr, D_, D_, D_, 1, 0, 0, 0, 0, 0, 0);

    // 7) Gate + output (float4 per lane)
    {
        int nquads = (int)(BND / 4);
        gate_out<<<dim3((nquads + 255) / 256, 1, 1), 256, 0, stream>>>(
            hout, xs, g1, g2, ubias, (float*)d_out, nquads);
    }

    // 8) Second tuple output: A passthrough
    hipMemcpyAsync((float*)d_out + BND, d_in[1],
                   (size_t)B_ * N_ * N_ * sizeof(float),
                   hipMemcpyDeviceToDevice, stream);
}
